// StateEncoder_5677946765417
// MI455X (gfx1250) — matrix operand order repaired and checked by reading, not compiled, not run on hardware
//
#include <hip/hip_runtime.h>
#include <cstddef>

typedef __attribute__((ext_vector_type(16))) _Float16 v16h;
typedef __attribute__((ext_vector_type(8)))  _Float16 v8h;
typedef __attribute__((ext_vector_type(8)))  float    v8f;

#define NB    16384
#define HB    120
#define BD    9
#define NS    32
#define NROWS (NB * BD)      /* 147456 */
#define NT16  (NROWS / 16)   /* 9216   */

__device__ __forceinline__ v8f wmma16(v16h a, v16h b, v8f c) {
  return __builtin_amdgcn_wmma_f32_16x16x32_f16(false, a, false, b, (short)0, c,
                                                false, false);
}
__device__ __forceinline__ float sigm(float x) { return 1.f / (1.f + __expf(-x)); }
__device__ __forceinline__ float phi_elu(float x) { return x > 0.f ? x + 1.f : __expf(x); }
__device__ __forceinline__ float softplusf(float x) {
  return x > 20.f ? x : log1pf(__expf(x));
}

// ---------------------------------------------------------------------------
// Pack a row-major W[Dout][Din] into WMMA f16 B-operand layout per 16x16x32
// tile: out[(n8*nK+ks)*32 + lane][i] = W[16*n8 + lane%16][32*ks + (lane/16)*8 + (i%8) + (i/8)*16]
// (zero padded out of range).  grid = nTiles*nK, block = 32.
// ---------------------------------------------------------------------------
__global__ __launch_bounds__(32) void pack_b_kernel(const float* __restrict__ W,
                                                    int Dout, int Din, int nK,
                                                    _Float16* __restrict__ out) {
  const int blk = blockIdx.x, L = threadIdx.x;
  const int n8 = blk / nK, ks = blk % nK;
  const int nIdx = 16 * n8 + (L & 15);
  _Float16* o = out + ((size_t)blk * 32 + L) * 16;
#pragma unroll
  for (int i = 0; i < 16; ++i) {
    const int k = 32 * ks + (L >> 4) * 8 + (i & 7) + (i >> 3) * 16;
    float v = (nIdx < Dout && k < Din) ? W[(size_t)nIdx * Din + k] : 0.f;
    o[i] = (_Float16)v;
  }
}

// ---------------------------------------------------------------------------
// Stage A: BatchNorm (eval) + lin_attn_2 (120 tokens x 9 dims) + LayerNorm,
// writes transposed [B*9][120].  One block of 128 threads per sample.
// ---------------------------------------------------------------------------
__global__ __launch_bounds__(128) void stageA_kernel(
    const float* __restrict__ x, const float* __restrict__ bn_g,
    const float* __restrict__ bn_b, const float* __restrict__ bn_m,
    const float* __restrict__ bn_v, const float* __restrict__ wq2,
    const float* __restrict__ wk2, const float* __restrict__ wv2,
    const float* __restrict__ wo2, const float* __restrict__ bo2,
    const float* __restrict__ ln2g, const float* __restrict__ ln2b,
    float* __restrict__ out) {
  __shared__ float swq[81], swk[81], swv[81], swo[81];
  __shared__ float kv[81], ksum[9];
  const int t = threadIdx.x;
  const size_t b = blockIdx.x;
  for (int i = t; i < 81; i += 128) {
    swq[i] = wq2[i]; swk[i] = wk2[i]; swv[i] = wv2[i]; swo[i] = wo2[i];
    kv[i] = 0.f;
  }
  if (t < 9) ksum[t] = 0.f;
  __syncthreads();
  float Q[9], K[9], V[9];
  if (t < HB) {
    float xb[9];
#pragma unroll
    for (int d = 0; d < 9; ++d) {
      const int idx = t * 9 + d;
      xb[d] = (x[b * (HB * BD) + idx] - bn_m[idx]) * rsqrtf(bn_v[idx] + 1e-5f) *
                  bn_g[idx] + bn_b[idx];
    }
#pragma unroll
    for (int e = 0; e < 9; ++e) {
      float sq = 0.f, sk = 0.f, sv = 0.f;
#pragma unroll
      for (int d = 0; d < 9; ++d) {
        sq += xb[d] * swq[e * 9 + d];
        sk += xb[d] * swk[e * 9 + d];
        sv += xb[d] * swv[e * 9 + d];
      }
      Q[e] = phi_elu(sq); K[e] = phi_elu(sk); V[e] = sv;
    }
#pragma unroll
    for (int d = 0; d < 9; ++d)
#pragma unroll
      for (int e = 0; e < 9; ++e) atomicAdd(&kv[d * 9 + e], K[d] * V[e]);
#pragma unroll
    for (int e = 0; e < 9; ++e) atomicAdd(&ksum[e], K[e]);
  }
  __syncthreads();
  if (t < HB) {
    float den = 0.f;
#pragma unroll
    for (int e = 0; e < 9; ++e) den += Q[e] * ksum[e];
    den = fmaxf(den, 1e-6f);
    float o[9];
#pragma unroll
    for (int e = 0; e < 9; ++e) {
      float num = 0.f;
#pragma unroll
      for (int d = 0; d < 9; ++d) num += Q[d] * kv[d * 9 + e];
      o[e] = num / den;
    }
    float y[9], mu = 0.f;
#pragma unroll
    for (int f = 0; f < 9; ++f) {
      float s = bo2[f] + o[f];
#pragma unroll
      for (int e = 0; e < 9; ++e) s += o[e] * swo[f * 9 + e];
      y[f] = s; mu += s;
    }
    mu *= (1.f / 9.f);
    float var = 0.f;
#pragma unroll
    for (int f = 0; f < 9; ++f) { float d = y[f] - mu; var += d * d; }
    var *= (1.f / 9.f);
    const float rs = rsqrtf(var + 1e-5f);
#pragma unroll
    for (int f = 0; f < 9; ++f)
      out[(b * 9 + f) * HB + t] = (y[f] - mu) * rs * ln2g[f] + ln2b[f];
  }
}

// ---------------------------------------------------------------------------
// Stage B: lin_attn_4 (S==1, d=120) via WMMA. One wave per 16 rows, in-place.
// ---------------------------------------------------------------------------
__device__ __forceinline__ void proj_wmma(const v16h* a,
                                          const _Float16* __restrict__ pack,
                                          bool doPhi, float (*dest)[128], int L) {
  const int row = L & 15, grp = L >> 4;
#pragma unroll
  for (int n8 = 0; n8 < 8; ++n8) {
    v8f acc = {};
#pragma unroll
    for (int ks = 0; ks < 4; ++ks) {
      v16h bm = *(const v16h*)(pack + ((size_t)(n8 * 4 + ks) * 32 + L) * 16);
      acc = wmma16(a[ks], bm, acc);
    }
#pragma unroll
    for (int g = 0; g < 8; ++g) {
      float v = acc[g];
      if (doPhi) v = phi_elu(v);
      dest[g + grp * 8][n8 * 16 + row] = v;
    }
  }
}

__global__ __launch_bounds__(32) void linattn4_kernel(
    float* __restrict__ buf, const _Float16* __restrict__ pQ,
    const _Float16* __restrict__ pK, const _Float16* __restrict__ pV,
    const _Float16* __restrict__ pO, const float* __restrict__ bo4,
    const float* __restrict__ ln4g, const float* __restrict__ ln4b) {
  __shared__ __align__(16) _Float16 x_h[16][128];
  __shared__ __align__(16) _Float16 o_h[16][128];
  __shared__ float qs[16][128];
  __shared__ float ks2[16][128];
  __shared__ float vs[16][128];
  const int L = threadIdx.x;
  const size_t base = (size_t)blockIdx.x * 16;
  for (int idx = L; idx < 16 * HB; idx += 32) {
    const int r = idx / HB, c = idx % HB;
    x_h[r][c] = (_Float16)buf[(base + r) * HB + c];
  }
  for (int idx = L; idx < 16 * 8; idx += 32)
    x_h[idx >> 3][HB + (idx & 7)] = (_Float16)0.f;
  __syncthreads();
  const int row = L & 15, grp = L >> 4;
  v16h a[4];
#pragma unroll
  for (int ks = 0; ks < 4; ++ks) {
    v8h lo = *(const v8h*)&x_h[row][32 * ks + grp * 8];
    v8h hi = *(const v8h*)&x_h[row][32 * ks + 16 + grp * 8];
    a[ks] = __builtin_shufflevector(lo, hi, 0, 1, 2, 3, 4, 5, 6, 7, 8, 9, 10, 11,
                                    12, 13, 14, 15);
  }
  proj_wmma(a, pQ, true, qs, L);
  proj_wmma(a, pK, true, ks2, L);
  proj_wmma(a, pV, false, vs, L);
  __syncthreads();
  {  // o = (Q.K) V / max(Q.K, 1e-6); lane pair (row, half) splits 120 cols
    const int c0 = 60 * grp;
    float qk = 0.f;
    for (int c = c0; c < c0 + 60; ++c) qk += qs[row][c] * ks2[row][c];
    qk += __shfl_xor(qk, 16);
    const float scale = qk / fmaxf(qk, 1e-6f);
    for (int c = c0; c < c0 + 60; ++c) {
      const float o = scale * vs[row][c];
      o_h[row][c] = (_Float16)o;
      qs[row][c] = o;  // keep f32 copy for residual
    }
  }
  for (int idx = L; idx < 16 * 8; idx += 32)
    o_h[idx >> 3][HB + (idx & 7)] = (_Float16)0.f;
  __syncthreads();
  v16h ao[4];
#pragma unroll
  for (int ks = 0; ks < 4; ++ks) {
    v8h lo = *(const v8h*)&o_h[row][32 * ks + grp * 8];
    v8h hi = *(const v8h*)&o_h[row][32 * ks + 16 + grp * 8];
    ao[ks] = __builtin_shufflevector(lo, hi, 0, 1, 2, 3, 4, 5, 6, 7, 8, 9, 10, 11,
                                     12, 13, 14, 15);
  }
#pragma unroll
  for (int n8 = 0; n8 < 8; ++n8) {
    v8f acc = {};
#pragma unroll
    for (int ks = 0; ks < 4; ++ks) {
      v16h bm = *(const v16h*)(pO + ((size_t)(n8 * 4 + ks) * 32 + L) * 16);
      acc = wmma16(ao[ks], bm, acc);
    }
    const int col = n8 * 16 + row;
    if (col < HB) {
#pragma unroll
      for (int g = 0; g < 8; ++g) {
        const int M = g + grp * 8;
        ks2[M][col] = acc[g] + bo4[col] + qs[M][col];  // out_proj + bias + residual
      }
    }
  }
  __syncthreads();
  {  // LayerNorm over 120, split across lane pair
    const int c0 = 60 * grp;
    float s = 0.f, s2 = 0.f;
    for (int c = c0; c < c0 + 60; ++c) { const float y = ks2[row][c]; s += y; s2 += y * y; }
    s += __shfl_xor(s, 16);
    s2 += __shfl_xor(s2, 16);
    const float mu = s / (float)HB;
    float var = s2 / (float)HB - mu * mu;
    var = fmaxf(var, 0.f);
    const float rs = rsqrtf(var + 1e-5f);
    for (int c = c0; c < c0 + 60; ++c)
      buf[(base + row) * HB + c] = (ks2[row][c] - mu) * rs * ln4g[c] + ln4b[c];
  }
}

// ---------------------------------------------------------------------------
// CGLSTM: one wave per 16 sequences, 120 steps. Per step: 6 gate WMMAs
// (o-gate is dead in the reference) + 2 contextual-gate WMMAs, K=32=NS.
// h round-trips through LDS (f16) between D-layout output and A-layout input;
// same-wave LDS ops are in-order on CDNA5.
// ---------------------------------------------------------------------------
__global__ __launch_bounds__(32) void cglstm_kernel(
    const float* __restrict__ xall, const _Float16* __restrict__ pWhh,
    const _Float16* __restrict__ pCg, const float* __restrict__ w_ih,
    const float* __restrict__ b_ih, const float* __restrict__ b_hh,
    const float* __restrict__ cg_w, const float* __restrict__ cg_ub,
    float* __restrict__ hout) {
  __shared__ __align__(16) _Float16 h_sh[16][32];
  __shared__ float x_sh[16][HB];
  const int L = threadIdx.x;
  const size_t seq0 = (size_t)blockIdx.x * 16;
  for (int idx = L; idx < 16 * HB; idx += 32) {
    const int r = idx / HB, c = idx % HB;
    x_sh[r][c] = xall[(seq0 + r) * HB + c];
  }
  for (int idx = L; idx < 16 * 32; idx += 32) ((_Float16*)h_sh)[idx] = (_Float16)0.f;
  __syncthreads();
  const int row = L & 15, grp = L >> 4;
  v16h bw[6], bc[2];
#pragma unroll
  for (int t = 0; t < 6; ++t) bw[t] = *(const v16h*)(pWhh + ((size_t)t * 32 + L) * 16);
#pragma unroll
  for (int q = 0; q < 2; ++q) bc[q] = *(const v16h*)(pCg + ((size_t)q * 32 + L) * 16);
  float biasg[6], wih_l[6], cgb[2], cgw_l[2];
#pragma unroll
  for (int t = 0; t < 6; ++t) {
    biasg[t] = b_ih[16 * t + row] + b_hh[16 * t + row];
    wih_l[t] = w_ih[16 * t + row];
  }
#pragma unroll
  for (int q = 0; q < 2; ++q) { cgb[q] = cg_ub[16 * q + row]; cgw_l[q] = cg_w[16 * q + row]; }
  float cst[2][8], hnew[2][8];
#pragma unroll
  for (int q = 0; q < 2; ++q)
#pragma unroll
    for (int g = 0; g < 8; ++g) { cst[q][g] = 0.f; hnew[q][g] = 0.f; }

  for (int t = 0; t < HB; ++t) {
    v8h lo = *(const v8h*)&h_sh[row][grp * 8];
    v8h hi = *(const v8h*)&h_sh[row][grp * 8 + 16];
    v16h a = __builtin_shufflevector(lo, hi, 0, 1, 2, 3, 4, 5, 6, 7, 8, 9, 10, 11,
                                     12, 13, 14, 15);
    float xv[8];
#pragma unroll
    for (int g = 0; g < 8; ++g) xv[g] = x_sh[g + grp * 8][t];
    float gi[2][8], gf[2][8], gg[2][8], gc[2][8];
#pragma unroll
    for (int q = 0; q < 2; ++q) {  // i gates: tiles 0,1
      v8f acc;
#pragma unroll
      for (int g = 0; g < 8; ++g) acc[g] = biasg[q] + xv[g] * wih_l[q];
      acc = wmma16(a, bw[q], acc);
#pragma unroll
      for (int g = 0; g < 8; ++g) gi[q][g] = sigm(acc[g]);
    }
#pragma unroll
    for (int q = 0; q < 2; ++q) {  // f gates: tiles 2,3
      v8f acc;
#pragma unroll
      for (int g = 0; g < 8; ++g) acc[g] = biasg[q + 2] + xv[g] * wih_l[q + 2];
      acc = wmma16(a, bw[q + 2], acc);
#pragma unroll
      for (int g = 0; g < 8; ++g) gf[q][g] = sigm(acc[g]);
    }
#pragma unroll
    for (int q = 0; q < 2; ++q) {  // g gates: tiles 4,5
      v8f acc;
#pragma unroll
      for (int g = 0; g < 8; ++g) acc[g] = biasg[q + 4] + xv[g] * wih_l[q + 4];
      acc = wmma16(a, bw[q + 4], acc);
#pragma unroll
      for (int g = 0; g < 8; ++g) gg[q][g] = tanhf(acc[g]);
    }
#pragma unroll
    for (int q = 0; q < 2; ++q) {  // contextual gate (uses old h)
      v8f acc;
#pragma unroll
      for (int g = 0; g < 8; ++g) acc[g] = cgb[q] + xv[g] * cgw_l[q];
      acc = wmma16(a, bc[q], acc);
#pragma unroll
      for (int g = 0; g < 8; ++g) gc[q][g] = sigm(acc[g]);
    }
#pragma unroll
    for (int q = 0; q < 2; ++q)
#pragma unroll
      for (int g = 0; g < 8; ++g) {
        const float cn = gf[q][g] * cst[q][g] + gi[q][g] * gg[q][g];
        cst[q][g] = cn;
        hnew[q][g] = gc[q][g] * tanhf(cn);
      }
#pragma unroll
    for (int q = 0; q < 2; ++q)
#pragma unroll
      for (int g = 0; g < 8; ++g)
        h_sh[g + grp * 8][16 * q + row] = (_Float16)hnew[q][g];
    __syncthreads();  // single-wave WG: S_NOP; LDS in-order gives the real guarantee
  }
#pragma unroll
  for (int q = 0; q < 2; ++q)
#pragma unroll
    for (int g = 0; g < 8; ++g)
      hout[(seq0 + g + grp * 8) * 32 + 16 * q + row] = hnew[q][g];
}

// ---------------------------------------------------------------------------
// Head: conv6 (32->60) + softplus, conv7 (60->15) + tanh, transpose_8, RevIN.
// One block of 160 threads per batch sample.
// ---------------------------------------------------------------------------
__global__ __launch_bounds__(160) void head_kernel(
    const float* __restrict__ h, const float* __restrict__ w6,
    const float* __restrict__ b6, const float* __restrict__ w7,
    const float* __restrict__ b7, const float* __restrict__ revw,
    const float* __restrict__ revb, float* __restrict__ enc) {
  __shared__ float t6[60];
  __shared__ float t7[135];
  __shared__ float stats[2];
  const int j = threadIdx.x;
  const size_t b = blockIdx.x;
  for (int v = 0; v < 9; ++v) {
    __syncthreads();
    if (j < 60) {
      const float* hr = h + (b * 9 + v) * 32;
      float s = b6[j];
#pragma unroll
      for (int k = 0; k < 32; ++k) s += hr[k] * w6[j * 32 + k];
      t6[j] = softplusf(s);
    }
    __syncthreads();
    if (j < 15) {
      float s = b7[j];
#pragma unroll
      for (int k = 0; k < 60; ++k) s += t6[k] * w7[j * 60 + k];
      t7[v * 15 + j] = tanhf(s);
    }
  }
  __syncthreads();
  if (j < 32) {
    float s = 0.f, s2 = 0.f;
    for (int k = j; k < 135; k += 32) { const float y = t7[k]; s += y; s2 += y * y; }
#pragma unroll
    for (int off = 16; off > 0; off >>= 1) {
      s += __shfl_down(s, off);
      s2 += __shfl_down(s2, off);
    }
    if (j == 0) {
      const float mu = s / 135.f;
      float var = (s2 - 135.f * mu * mu) / 134.f;  // ddof=1
      var = fmaxf(var, 0.f);
      stats[0] = mu;
      stats[1] = fmaxf(sqrtf(var), 1e-5f);
    }
  }
  __syncthreads();
  if (j < 135) {
    const int bd = j / 15, nf = j % 15;           // transpose_8
    const float y = t7[nf * 9 + bd];
    enc[b * 135 + j] = (y - stats[0]) / stats[1] * revw[j] + revb[j];
  }
}

// ---------------------------------------------------------------------------
extern "C" void kernel_launch(void* const* d_in, const int* in_sizes, int n_in,
                              void* d_out, int out_size, void* d_ws,
                              size_t ws_size, hipStream_t stream) {
  (void)in_sizes; (void)n_in; (void)out_size; (void)ws_size;
  const float* x    = (const float*)d_in[0];
  const float* bn_g = (const float*)d_in[1];
  const float* bn_b = (const float*)d_in[2];
  const float* bn_m = (const float*)d_in[3];
  const float* bn_v = (const float*)d_in[4];
  const float* wq2  = (const float*)d_in[5];
  const float* wk2  = (const float*)d_in[6];
  const float* wv2  = (const float*)d_in[7];
  const float* wo2  = (const float*)d_in[8];
  const float* bo2  = (const float*)d_in[9];
  const float* ln2g = (const float*)d_in[10];
  const float* ln2b = (const float*)d_in[11];
  const float* wq4  = (const float*)d_in[12];
  const float* wk4  = (const float*)d_in[13];
  const float* wv4  = (const float*)d_in[14];
  const float* wo4  = (const float*)d_in[15];
  const float* bo4  = (const float*)d_in[16];
  const float* ln4g = (const float*)d_in[17];
  const float* ln4b = (const float*)d_in[18];
  const float* w_ih = (const float*)d_in[19];
  const float* w_hh = (const float*)d_in[20];
  const float* b_ih = (const float*)d_in[21];
  const float* b_hh = (const float*)d_in[22];
  const float* cg_w = (const float*)d_in[23];
  const float* cg_uw= (const float*)d_in[24];
  const float* cg_ub= (const float*)d_in[25];
  const float* w6   = (const float*)d_in[26];
  const float* b6   = (const float*)d_in[27];
  const float* w7   = (const float*)d_in[28];
  const float* b7   = (const float*)d_in[29];
  const float* revw = (const float*)d_in[30];
  const float* revb = (const float*)d_in[31];

  char* ws = (char*)d_ws;
  size_t off = 0;
  float* buf1 = (float*)(ws + off);
  off += (size_t)NROWS * HB * sizeof(float);           // 70,778,880 (256-aligned)
  _Float16* pQ = (_Float16*)(ws + off);   off += 8 * 4 * 32 * 16 * sizeof(_Float16);
  _Float16* pK = (_Float16*)(ws + off);   off += 8 * 4 * 32 * 16 * sizeof(_Float16);
  _Float16* pV = (_Float16*)(ws + off);   off += 8 * 4 * 32 * 16 * sizeof(_Float16);
  _Float16* pO = (_Float16*)(ws + off);   off += 8 * 4 * 32 * 16 * sizeof(_Float16);
  _Float16* pWhh = (_Float16*)(ws + off); off += 8 * 32 * 16 * sizeof(_Float16);
  _Float16* pCg = (_Float16*)(ws + off);  off += 2 * 32 * 16 * sizeof(_Float16);

  float* hout = (float*)d_out;                         // [B, 288] == [B*9, 32]
  float* enc  = (float*)d_out + (size_t)NB * (BD * NS);

  // Pack weights into WMMA B-operand layout (f16).
  pack_b_kernel<<<dim3(8 * 4), dim3(32), 0, stream>>>(wq4, HB, HB, 4, pQ);
  pack_b_kernel<<<dim3(8 * 4), dim3(32), 0, stream>>>(wk4, HB, HB, 4, pK);
  pack_b_kernel<<<dim3(8 * 4), dim3(32), 0, stream>>>(wv4, HB, HB, 4, pV);
  pack_b_kernel<<<dim3(8 * 4), dim3(32), 0, stream>>>(wo4, HB, HB, 4, pO);
  pack_b_kernel<<<dim3(8), dim3(32), 0, stream>>>(w_hh, 4 * NS, NS, 1, pWhh);
  pack_b_kernel<<<dim3(2), dim3(32), 0, stream>>>(cg_uw, NS, NS, 1, pCg);

  stageA_kernel<<<dim3(NB), dim3(128), 0, stream>>>(
      x, bn_g, bn_b, bn_m, bn_v, wq2, wk2, wv2, wo2, bo2, ln2g, ln2b, buf1);

  linattn4_kernel<<<dim3(NT16), dim3(32), 0, stream>>>(
      buf1, pQ, pK, pV, pO, bo4, ln4g, ln4b);

  cglstm_kernel<<<dim3(NT16), dim3(32), 0, stream>>>(
      buf1, pWhh, pCg, w_ih, b_ih, b_hh, cg_w, cg_ub, hout);

  head_kernel<<<dim3(NB), dim3(160), 0, stream>>>(
      hout, w6, b6, w7, b7, revw, revb, enc);
}
